// _recur_58901181497902
// MI455X (gfx1250) — matrix operand order repaired and checked by reading, not compiled, not run on hardware
//
#include <hip/hip_runtime.h>
#include <cstdint>
#include <cstddef>

// Problem constants (B, T, DIN, H, O) = (8, 128, 64, 256, 64)
#define B_   8
#define T_   128
#define DIN_ 64
#define H_   256
#define O_   64

typedef _Float16 v16h __attribute__((ext_vector_type(16)));
typedef float    v8f  __attribute__((ext_vector_type(8)));

__device__ __forceinline__ float sigm(float x) { return 1.0f / (1.0f + expf(-x)); }

// ---------------------------------------------------------------------------
// A-fragment loader: 16x32 f16 A tile, per CDNA5 ISA 7.12.2 layout.
// lane L: m = L%16, g = L/16; element e: e<8 -> k = 8g+e ; e>=8 -> k = 16+8g+(e-8)
// => two contiguous 8-float runs at (k0+8g) and (k0+16+8g). Convert f32->f16.
// ---------------------------------------------------------------------------
__device__ __forceinline__ v16h load_a_frag_f32(const float* __restrict__ A, int lda,
                                                int row_base, int k0, int lane) {
  const int m = lane & 15, g = lane >> 4;
  const float* p = A + (size_t)(row_base + m) * lda + (k0 + 8 * g);
  float t[16];
  *(float4*)(t + 0)  = *(const float4*)(p + 0);
  *(float4*)(t + 4)  = *(const float4*)(p + 4);
  *(float4*)(t + 8)  = *(const float4*)(p + 16);
  *(float4*)(t + 12) = *(const float4*)(p + 20);
  v16h a;
#pragma unroll
  for (int e = 0; e < 16; ++e) a[e] = (_Float16)t[e];
  return a;
}

// ---------------------------------------------------------------------------
// Weight pre-swizzle: W (K x N, f32 row-major) -> f16 fragment-ready layout.
// B-fragment (K x 16 slab): lane L holds column n = tn*16 + L%16, g = L/16,
// element e -> k = kb*32 + (e<8 ? 8*g+e : 16+8*g+(e-8)).  Stored so a wave reads its fragment as
// one contiguous 32-byte v16h load: Bsw[((tn*KB + kb)*32 + lane)*16 + e].
// ---------------------------------------------------------------------------
__global__ void swizzle_b_kernel(const float* __restrict__ W, int K, int N,
                                 _Float16* __restrict__ Bsw) {
  const int total = K * N;
  int idx = blockIdx.x * blockDim.x + threadIdx.x;
  if (idx >= total) return;
  const int e    = idx & 15;
  const int lane = (idx >> 4) & 31;
  const int blk  = idx >> 9;           // tn*KB + kb
  const int KB   = K >> 5;
  const int kb   = blk % KB;
  const int tn   = blk / KB;
  const int g    = lane >> 4;
  const int n    = tn * 16 + (lane & 15);
  const int k    = kb * 32 + (e < 8 ? 8 * g + e : 16 + 8 * g + (e - 8));
  Bsw[idx] = (_Float16)W[(size_t)k * N + n];
}

// ---------------------------------------------------------------------------
// Fused encoder+decoder LSTM. One block per batch (independent). h in LDS,
// c per-thread. Pure VALU: ~1.4 GFLOP total, weights L2-resident.
// ---------------------------------------------------------------------------
__global__ __launch_bounds__(256) void lstm_kernel(
    const float* __restrict__ x,
    const float* __restrict__ eWx, const float* __restrict__ eWh, const float* __restrict__ eb,
    const float* __restrict__ dWx, const float* __restrict__ dWh, const float* __restrict__ db,
    float* __restrict__ F) {
  const int b = blockIdx.x, j = threadIdx.x;
  __shared__ float hs[H_];
  __shared__ float feat[2 * DIN_];
  float c = 0.f;
  hs[j] = 0.f;
  __syncthreads();
  // encoder: 128 steps
  for (int t = 0; t < T_; ++t) {
    if (j < DIN_) feat[j] = x[((size_t)b * T_ + t) * DIN_ + j];
    __syncthreads();
    float zi = eb[j], zf = eb[H_ + j], zg = eb[2 * H_ + j], zo = eb[3 * H_ + j];
    for (int k = 0; k < DIN_; ++k) {
      const float xv = feat[k];
      const float* wr = eWx + (size_t)k * (4 * H_);
      zi += xv * wr[j]; zf += xv * wr[H_ + j]; zg += xv * wr[2 * H_ + j]; zo += xv * wr[3 * H_ + j];
    }
    for (int k = 0; k < H_; ++k) {
      const float hv = hs[k];
      const float* wr = eWh + (size_t)k * (4 * H_);
      zi += hv * wr[j]; zf += hv * wr[H_ + j]; zg += hv * wr[2 * H_ + j]; zo += hv * wr[3 * H_ + j];
    }
    c = sigm(zf) * c + sigm(zi) * tanhf(zg);
    const float hn = sigm(zo) * tanhf(c);
    __syncthreads();
    hs[j] = hn;
    __syncthreads();
  }
  // decoder: 127 steps, input = concat(x[:, t], x[:, 127-t]); writes F
  for (int t = 0; t < T_ - 1; ++t) {
    if (j < 2 * DIN_) {
      feat[j] = (j < DIN_) ? x[((size_t)b * T_ + t) * DIN_ + j]
                           : x[((size_t)b * T_ + (T_ - 1 - t)) * DIN_ + (j - DIN_)];
    }
    __syncthreads();
    float zi = db[j], zf = db[H_ + j], zg = db[2 * H_ + j], zo = db[3 * H_ + j];
    for (int k = 0; k < 2 * DIN_; ++k) {
      const float xv = feat[k];
      const float* wr = dWx + (size_t)k * (4 * H_);
      zi += xv * wr[j]; zf += xv * wr[H_ + j]; zg += xv * wr[2 * H_ + j]; zo += xv * wr[3 * H_ + j];
    }
    for (int k = 0; k < H_; ++k) {
      const float hv = hs[k];
      const float* wr = dWh + (size_t)k * (4 * H_);
      zi += hv * wr[j]; zf += hv * wr[H_ + j]; zg += hv * wr[2 * H_ + j]; zo += hv * wr[3 * H_ + j];
    }
    c = sigm(zf) * c + sigm(zi) * tanhf(zg);
    const float hn = sigm(zo) * tanhf(c);
    __syncthreads();
    hs[j] = hn;
    F[((size_t)b * (T_ - 1) + t) * H_ + j] = hn;
    __syncthreads();
  }
}

// D mid row (tanh(m)) + zero the GEMM pad rows of F and D.
__global__ void prep_kernel(const float* __restrict__ m, float* __restrict__ D,
                            float* __restrict__ F) {
  const int tid = blockIdx.x * blockDim.x + threadIdx.x;   // 2048 threads
  if (tid < B_ * H_) {
    const int b = tid >> 8, h = tid & (H_ - 1);
    D[((size_t)b * 255 + 127) * H_ + h] = tanhf(m[h]);
    F[(size_t)1016 * H_ + tid] = 0.f;   // F pad rows 1016..1023
    D[(size_t)2040 * H_ + tid] = 0.f;   // D pad rows 2040..2047
  }
}

// ---------------------------------------------------------------------------
// Generic WMMA GEMM: C[row, col] = act(A(f32) @ Wsw(f16) + bias).
// One 16x16 output tile per wave-iteration, K in 32-wide WMMA steps.
// ---------------------------------------------------------------------------
template <int ACT>   // 0 = none, 1 = tanh
__global__ __launch_bounds__(256) void gemm_bias_kernel(
    const float* __restrict__ A, int lda,
    const _Float16* __restrict__ Bsw, int KB,
    const float* __restrict__ bias,
    float* __restrict__ C, int ldc, int M, int MT, int NT) {
  const int lane = threadIdx.x & 31;
  const int wglb = (blockIdx.x * blockDim.x + threadIdx.x) >> 5;
  const int nw   = (gridDim.x * blockDim.x) >> 5;
  const int n = lane & 15, g = lane >> 4;
  for (int tile = wglb; tile < MT * NT; tile += nw) {
    const int tm = tile % MT, tn = tile / MT;
    v8f acc = {};
    const _Float16* bp = Bsw + ((size_t)tn * KB * 32 + lane) * 16;
#pragma unroll 2
    for (int kb = 0; kb < KB; ++kb) {
      v16h a  = load_a_frag_f32(A, lda, tm * 16, kb * 32, lane);
      v16h bf = *(const v16h*)(bp + (size_t)kb * 512);
      acc = __builtin_amdgcn_wmma_f32_16x16x32_f16(false, a, false, bf, (short)0, acc,
                                                   false, false);
    }
    const int col = tn * 16 + n;
    const float bv = bias[col];
#pragma unroll
    for (int r = 0; r < 8; ++r) {
      const int row = tm * 16 + r + 8 * g;        // C layout: VGPR r -> M = r + 8g
      if (row < M) {
        float v = acc[r] + bv;
        if (ACT == 1) v = tanhf(v);
        C[(size_t)row * ldc + col] = v;
      }
    }
  }
}

// ---------------------------------------------------------------------------
// Dual-gate GEMM for phase C:  out = tanh(F@Wlin+b1) * sigmoid(F@Wgate+b2),
// scattered into D either reversed (rows 126-t) or forward (rows 128+t).
// A fragments are shared between the two WMMA accumulators.
// ---------------------------------------------------------------------------
__global__ __launch_bounds__(256) void dual_gate_kernel(
    const float* __restrict__ F,
    const _Float16* __restrict__ BswLin, const _Float16* __restrict__ BswGate,
    const float* __restrict__ bLin, const float* __restrict__ bGate,
    float* __restrict__ D, int reverse) {
  const int lane = threadIdx.x & 31;
  const int wglb = (blockIdx.x * blockDim.x + threadIdx.x) >> 5;
  const int nw   = (gridDim.x * blockDim.x) >> 5;
  const int n = lane & 15, g = lane >> 4;
  for (int tile = wglb; tile < 64 * 16; tile += nw) {   // MT=64 (1024 pad rows), NT=16
    const int tm = tile & 63, tn = tile >> 6;
    v8f acc1 = {}, acc2 = {};
    const _Float16* bp1 = BswLin  + ((size_t)tn * 8 * 32 + lane) * 16;
    const _Float16* bp2 = BswGate + ((size_t)tn * 8 * 32 + lane) * 16;
#pragma unroll 2
    for (int kb = 0; kb < 8; ++kb) {
      v16h a  = load_a_frag_f32(F, H_, tm * 16, kb * 32, lane);
      v16h b1 = *(const v16h*)(bp1 + (size_t)kb * 512);
      v16h b2 = *(const v16h*)(bp2 + (size_t)kb * 512);
      acc1 = __builtin_amdgcn_wmma_f32_16x16x32_f16(false, a, false, b1, (short)0, acc1,
                                                    false, false);
      acc2 = __builtin_amdgcn_wmma_f32_16x16x32_f16(false, a, false, b2, (short)0, acc2,
                                                    false, false);
    }
    const int col = tn * 16 + n;
    const float b1v = bLin[col], b2v = bGate[col];
#pragma unroll
    for (int r = 0; r < 8; ++r) {
      const int row = tm * 16 + r + 8 * g;
      if (row < B_ * (T_ - 1)) {                  // 1016 real rows
        const int bb = row / (T_ - 1), t = row % (T_ - 1);
        const int orow = bb * 255 + (reverse ? (126 - t) : (128 + t));
        D[(size_t)orow * H_ + col] = tanhf(acc1[r] + b1v) * sigm(acc2[r] + b2v);
      }
    }
  }
}

// ---------------------------------------------------------------------------
// Persistent cell kernel helpers (512 threads = 16 waves per workgroup)
// ---------------------------------------------------------------------------

// xl = tanh(A(128x256, f32 global) @ Wsw(f16) + xb) -> LDS. 128 tiles / 16 waves.
__device__ __forceinline__ void gemm_to_xl(const float* __restrict__ A,
                                           const _Float16* __restrict__ Bsw,
                                           const float* __restrict__ bias,
                                           float* __restrict__ xl, int tid) {
  const int lane = tid & 31, wave = tid >> 5;
  const int n = lane & 15, g = lane >> 4;
#pragma unroll 1
  for (int tt = 0; tt < 8; ++tt) {
    const int tile = wave * 8 + tt;
    const int tm = tile & 7, tn = tile >> 3;      // 8 M-tiles x 16 N-tiles
    v8f acc = {};
    const _Float16* bp = Bsw + ((size_t)tn * 8 * 32 + lane) * 16;
#pragma unroll 2
    for (int kb = 0; kb < 8; ++kb) {
      v16h a  = load_a_frag_f32(A, H_, tm * 16, kb * 32, lane);
      v16h bf = *(const v16h*)(bp + (size_t)kb * 512);
      acc = __builtin_amdgcn_wmma_f32_16x16x32_f16(false, a, false, bf, (short)0, acc,
                                                   false, false);
    }
    const int col = tn * 16 + n;
    const float bv = bias[col];
#pragma unroll
    for (int r = 0; r < 8; ++r) {
      const int row = tm * 16 + r + 8 * g;
      xl[row * H_ + col] = tanhf(acc[r] + bv);
    }
  }
}

// Banded correlation: out[i,h] = sum_j xl[j,h] * CD[127-i+j, h], with per-gate
// epilogue. CD streamed through LDS in 64-channel chunks (L2 -> 64 KB LDS).
template <int GATE>   // 0: u=sigmoid -> ubuf ; 1: zh=sigmoid*hid ; 2: hid update
__device__ __forceinline__ void corr_gate(const float* __restrict__ xl,
                                          float* __restrict__ cdbuf,
                                          const float* __restrict__ CD,
                                          const float* __restrict__ gbias,
                                          float* __restrict__ hidb,
                                          float* __restrict__ ubufb,
                                          float* __restrict__ zhb, int tid) {
  for (int hc = 0; hc < H_; hc += 64) {
    for (int idx = tid; idx < 255 * 64; idx += 512) {
      const int jj = idx >> 6, hl = idx & 63;
      cdbuf[idx] = CD[(size_t)jj * H_ + hc + hl];
    }
    __syncthreads();
    const int i   = tid >> 2;            // 0..127
    const int hl0 = (tid & 3) << 4;      // 0,16,32,48
    float acc[16];
#pragma unroll
    for (int q = 0; q < 16; ++q) acc[q] = 0.f;
    const int off = 127 - i;
#pragma unroll 1
    for (int j = 0; j < T_; ++j) {
      const float* xr = xl + j * H_ + hc + hl0;
      const float* cr = cdbuf + (j + off) * 64 + hl0;
#pragma unroll
      for (int q = 0; q < 4; ++q) {
        const float4 xv = *(const float4*)(xr + 4 * q);
        const float4 cv = *(const float4*)(cr + 4 * q);
        acc[4 * q + 0] += xv.x * cv.x;
        acc[4 * q + 1] += xv.y * cv.y;
        acc[4 * q + 2] += xv.z * cv.z;
        acc[4 * q + 3] += xv.w * cv.w;
      }
    }
    const int hbase = hc + hl0;
#pragma unroll
    for (int q = 0; q < 16; ++q) {
      const int h = hbase + q;
      const size_t o = (size_t)i * H_ + h;
      const float v = acc[q] + gbias[h];
      if (GATE == 0) {
        ubufb[o] = sigm(v);
      } else if (GATE == 1) {
        zhb[o] = sigm(v) * hidb[o];
      } else {
        const float cand = tanhf(v);
        const float u = ubufb[o];
        hidb[o] = u * hidb[o] + (1.f - u) * cand;
      }
    }
    __syncthreads();
  }
}

// Persistent recurrent-cell kernel: one workgroup per batch, 128 scan steps.
// Dynamic LDS = 128*256 (xl) + 255*64 (cdbuf) floats = 192 KB of the 320 KB WGP LDS.
__global__ __launch_bounds__(512) void cell_kernel(
    const float* __restrict__ u_xb, const float* __restrict__ z_xb,
    const float* __restrict__ c_xb,
    const float* __restrict__ u_bias, const float* __restrict__ z_bias,
    const float* __restrict__ c_bias,
    const _Float16* __restrict__ BswU, const _Float16* __restrict__ BswZ,
    const _Float16* __restrict__ BswC,
    const float* __restrict__ CU, const float* __restrict__ CZ,
    const float* __restrict__ CC,
    float* __restrict__ hid, float* __restrict__ zh, float* __restrict__ ubuf) {
  extern __shared__ float smem[];
  float* xl    = smem;                 // 128*256 f32
  float* cdbuf = smem + T_ * H_;       // 255*64  f32
  const int b = blockIdx.x, tid = threadIdx.x;
  float* hidb = hid + (size_t)b * T_ * H_;
  float* zhb  = zh  + (size_t)b * T_ * H_;
  float* ubb  = ubuf + (size_t)b * T_ * H_;
  const float* CUb = CU + (size_t)b * 255 * H_;
  const float* CZb = CZ + (size_t)b * 255 * H_;
  const float* CCb = CC + (size_t)b * 255 * H_;
  for (int it = 0; it < T_; ++it) {
    gemm_to_xl(hidb, BswU, u_xb, xl, tid);
    __syncthreads();
    corr_gate<0>(xl, cdbuf, CUb, u_bias, hidb, ubb, zhb, tid);
    gemm_to_xl(hidb, BswZ, z_xb, xl, tid);
    __syncthreads();
    corr_gate<1>(xl, cdbuf, CZb, z_bias, hidb, ubb, zhb, tid);
    gemm_to_xl(zhb, BswC, c_xb, xl, tid);
    __syncthreads();
    corr_gate<2>(xl, cdbuf, CCb, c_bias, hidb, ubb, zhb, tid);
  }
}

// ---------------------------------------------------------------------------
// Host launcher
// ---------------------------------------------------------------------------
extern "C" void kernel_launch(void* const* d_in, const int* in_sizes, int n_in,
                              void* d_out, int out_size, void* d_ws, size_t ws_size,
                              hipStream_t stream) {
  (void)in_sizes; (void)n_in; (void)out_size; (void)ws_size;
  const float* x          = (const float*)d_in[0];
  const float* enc_Wx     = (const float*)d_in[1];
  const float* enc_Wh     = (const float*)d_in[2];
  const float* enc_b      = (const float*)d_in[3];
  const float* dec_Wx     = (const float*)d_in[4];
  const float* dec_Wh     = (const float*)d_in[5];
  const float* dec_b      = (const float*)d_in[6];
  const float* rev_lin_k  = (const float*)d_in[7];
  const float* rev_lin_b  = (const float*)d_in[8];
  const float* rev_gate_k = (const float*)d_in[9];
  const float* rev_gate_b = (const float*)d_in[10];
  const float* for_lin_k  = (const float*)d_in[11];
  const float* for_lin_b  = (const float*)d_in[12];
  const float* for_gate_k = (const float*)d_in[13];
  const float* for_gate_b = (const float*)d_in[14];
  const float* m          = (const float*)d_in[15];
  const float* u_wk  = (const float*)d_in[16];
  const float* u_wb  = (const float*)d_in[17];
  const float* u_xk  = (const float*)d_in[18];
  const float* u_xb  = (const float*)d_in[19];
  const float* u_bias= (const float*)d_in[20];
  const float* z_wk  = (const float*)d_in[21];
  const float* z_wb  = (const float*)d_in[22];
  const float* z_xk  = (const float*)d_in[23];
  const float* z_xb  = (const float*)d_in[24];
  const float* z_bias= (const float*)d_in[25];
  const float* c_wk  = (const float*)d_in[26];
  const float* c_wb  = (const float*)d_in[27];
  const float* c_xk  = (const float*)d_in[28];
  const float* c_xb  = (const float*)d_in[29];
  const float* c_bias= (const float*)d_in[30];
  const float* h_lin_k = (const float*)d_in[31];
  const float* h_lin_b = (const float*)d_in[32];
  const float* o_lin_k = (const float*)d_in[33];
  const float* o_lin_b = (const float*)d_in[34];

  // Workspace carve-out (256B aligned slabs), ~13.3 MB total
  char* base = (char*)d_ws;
  size_t off = 0;
  auto af = [&](size_t elems) {
    float* p = (float*)(base + off);
    off += (elems * sizeof(float) + 255) & ~(size_t)255;
    return p;
  };
  auto ah = [&](size_t elems) {
    _Float16* p = (_Float16*)(base + off);
    off += (elems * sizeof(_Float16) + 255) & ~(size_t)255;
    return p;
  };
  float* F   = af((size_t)1024 * H_);   // B*127 real rows, padded to 1024
  float* D   = af((size_t)2048 * H_);   // B*255 real rows, padded to 2048
  float* CU  = af((size_t)2048 * H_);
  float* CZ  = af((size_t)2048 * H_);
  float* CC  = af((size_t)2048 * H_);
  float* hid = af((size_t)B_ * T_ * H_);
  float* zh  = af((size_t)B_ * T_ * H_);
  float* ub  = af((size_t)B_ * T_ * H_);
  _Float16* BswRevLin  = ah((size_t)H_ * H_);
  _Float16* BswRevGate = ah((size_t)H_ * H_);
  _Float16* BswForLin  = ah((size_t)H_ * H_);
  _Float16* BswForGate = ah((size_t)H_ * H_);
  _Float16* BswUwk = ah((size_t)H_ * H_);
  _Float16* BswZwk = ah((size_t)H_ * H_);
  _Float16* BswCwk = ah((size_t)H_ * H_);
  _Float16* BswUxk = ah((size_t)H_ * H_);
  _Float16* BswZxk = ah((size_t)H_ * H_);
  _Float16* BswCxk = ah((size_t)H_ * H_);
  _Float16* BswHlin = ah((size_t)DIN_ * H_);
  _Float16* BswOlin = ah((size_t)H_ * O_);

  auto sw = [&](const float* W, int K, int N, _Float16* Bsw) {
    const int total = K * N;
    swizzle_b_kernel<<<(total + 255) / 256, 256, 0, stream>>>(W, K, N, Bsw);
  };
  sw(rev_lin_k, H_, H_, BswRevLin);
  sw(rev_gate_k, H_, H_, BswRevGate);
  sw(for_lin_k, H_, H_, BswForLin);
  sw(for_gate_k, H_, H_, BswForGate);
  sw(u_wk, H_, H_, BswUwk);
  sw(z_wk, H_, H_, BswZwk);
  sw(c_wk, H_, H_, BswCwk);
  sw(u_xk, H_, H_, BswUxk);
  sw(z_xk, H_, H_, BswZxk);
  sw(c_xk, H_, H_, BswCxk);
  sw(h_lin_k, DIN_, H_, BswHlin);
  sw(o_lin_k, H_, O_, BswOlin);

  // Phase A/B: encoder+decoder LSTM -> F
  lstm_kernel<<<B_, 256, 0, stream>>>(x, enc_Wx, enc_Wh, enc_b, dec_Wx, dec_Wh, dec_b, F);
  // D mid row + pad-row zeroing
  prep_kernel<<<8, 256, 0, stream>>>(m, D, F);
  // Phase C: D rows from F (reversed R, forward F1)
  dual_gate_kernel<<<128, 256, 0, stream>>>(F, BswRevLin, BswRevGate, rev_lin_b, rev_gate_b, D, 1);
  dual_gate_kernel<<<128, 256, 0, stream>>>(F, BswForLin, BswForGate, for_lin_b, for_gate_b, D, 0);
  // Phase D: CU/CZ/CC = D @ wk + wb (dconv factored through the gather)
  gemm_bias_kernel<0><<<256, 256, 0, stream>>>(D, H_, BswUwk, 8, u_wb, CU, H_, 2048, 128, 16);
  gemm_bias_kernel<0><<<256, 256, 0, stream>>>(D, H_, BswZwk, 8, z_wb, CZ, H_, 2048, 128, 16);
  gemm_bias_kernel<0><<<256, 256, 0, stream>>>(D, H_, BswCwk, 8, c_wb, CC, H_, 2048, 128, 16);
  // Phase E: hidden0 = tanh(x @ h_lin_k + b)
  gemm_bias_kernel<1><<<128, 256, 0, stream>>>(x, DIN_, BswHlin, 2, h_lin_b, hid, H_, 1024, 64, 16);

  // Phase F: 128-step persistent recurrence, one WGP per batch, 192 KB LDS.
  const size_t smem = (size_t)(T_ * H_ + 255 * 64) * sizeof(float);  // 196,352 B
  hipFuncSetAttribute((const void*)cell_kernel,
                      hipFuncAttributeMaxDynamicSharedMemorySize, (int)smem);
  cell_kernel<<<B_, 512, smem, stream>>>(u_xb, z_xb, c_xb, u_bias, z_bias, c_bias,
                                         BswUxk, BswZxk, BswCxk, CU, CZ, CC,
                                         hid, zh, ub);

  // Phase G: out = hidden @ o_lin_k + o_lin_b
  gemm_bias_kernel<0><<<32, 256, 0, stream>>>(hid, H_, BswOlin, 8, o_lin_b,
                                              (float*)d_out, O_, 1024, 64, 4);
}
